// LinkPredictionSRHGN_36644660969558
// MI455X (gfx1250) — matrix operand order repaired and checked by reading, not compiled, not run on hardware
//
#include <hip/hip_runtime.h>

// ---------------------------------------------------------------------------
// SRHGN link prediction for MI455X (gfx1250, wave32, WMMA).
// Dense 128x128 projections: v_wmma_f32_16x16x32_bf16 with
//   - weights pre-transposed/converted to bf16 (contiguous 32B B-fragments)
//   - A tile staged once per block in LDS as bf16 (8x wave reuse)
// Edge softmax/aggregation: HBM-streaming atomic kernels (roofline bottleneck).
// ---------------------------------------------------------------------------

#define DFEAT 128
#define HN 4            // node heads
#define HT 4            // type heads
#define NAUT 100000
#define NPAP 100000
#define NEDGE 500000
#define NPAIR 100000

typedef __attribute__((ext_vector_type(16))) __bf16 v16bf;
typedef __attribute__((ext_vector_type(8)))  __bf16 v8bf;
typedef __attribute__((ext_vector_type(8)))  float  v8f;

__device__ __forceinline__ float gelu_exact(float x) {
    return 0.5f * x * (1.0f + erff(x * 0.7071067811865476f));
}
__device__ __forceinline__ float lrelu(float x) { return x > 0.0f ? x : 0.01f * x; }

// Monotonic float<->int map so segment-max can use integer atomicMax.
__device__ __forceinline__ int f2ord(float f) {
    int i = __float_as_int(f);
    return i >= 0 ? i : (i ^ 0x7fffffff);
}
__device__ __forceinline__ float ord2f(int k) {
    int i = k >= 0 ? k : (k ^ 0x7fffffff);
    return __int_as_float(i);
}

// ---------------------------------------------------------------------------
// Fills
// ---------------------------------------------------------------------------
__global__ void fill_f32(float* p, float v, long n) {
    long i = (long)blockIdx.x * blockDim.x + threadIdx.x;
    if (i < n) p[i] = v;
}
__global__ void fill_i32(int* p, int v, long n) {
    long i = (long)blockIdx.x * blockDim.x + threadIdx.x;
    if (i < n) p[i] = v;
}

// ---------------------------------------------------------------------------
// One-shot weight prep: Wt[n*128+k] = bf16(W[k*128+n])  (transpose + convert)
// After this, a WMMA B-fragment for lane n is 16 *consecutive* bf16 values.
// ---------------------------------------------------------------------------
__global__ void conv_w_t(const float* __restrict__ W, __bf16* __restrict__ Wt)
{
    int idx = blockIdx.x * blockDim.x + threadIdx.x;   // 16384
    int n = idx >> 7, k = idx & 127;
    Wt[idx] = (__bf16)W[k * DFEAT + n];
}

// ---------------------------------------------------------------------------
// WMMA GEMM: C[N,128] = act(A[N,128] @ W[128,128] + b[128]), W given as
// bf16 transposed (Wt[n][k]).
// block = 256 threads = 8 waves; block owns a 16-row stripe; wave w owns
// output cols [16w,16w+16). A tile staged in LDS as bf16 (4KB, 8x reuse).
// Fragment layouts used here (wave32; A/B k order is contiguous, not the ISA table):
//   A 16x32 bf16 : lane -> M = lane&15, half = lane>>4;
//                  v16bf elems 0-15 = K: k0 + 16*half + 0..15
//   B 32x16 bf16 : lane -> N = lane&15; v16bf elems = K: k0+16*half+0..15
//   C 16x16 f32  : vgpr r -> row = r + 8*half, col = lane&15
// ---------------------------------------------------------------------------
__global__ __launch_bounds__(256) void gemm128_wmma(
    const float* __restrict__ A, const __bf16* __restrict__ Wt,
    const float* __restrict__ bias, float* __restrict__ C, int act)
{
    __shared__ __bf16 As[16 * DFEAT];     // 4KB bf16 A tile

    const int tid  = threadIdx.x;
    const int lane = tid & 31;
    const int wave = tid >> 5;            // 0..7 -> 16-col tile
    const int half = lane >> 4;
    const int mn   = lane & 15;
    const long row0 = (long)blockIdx.x * 16;
    const int  col0 = wave * 16;

    // ---- stage A tile: 16 rows x 128 f32 -> bf16 in LDS (coalesced) ----
    {
        const int r = tid >> 4;           // 0..15
        const int c = (tid & 15) * 8;     // 0..120
        const float* src = A + (row0 + r) * DFEAT + c;
        v8bf t;
#pragma unroll
        for (int i = 0; i < 8; ++i) t[i] = (__bf16)src[i];
        *(v8bf*)(As + r * DFEAT + c) = t;
    }
    __syncthreads();

    v8f acc;
#pragma unroll
    for (int i = 0; i < 8; ++i) acc[i] = 0.0f;

#pragma unroll
    for (int kb = 0; kb < 4; ++kb) {
        const int k0 = kb * 32;
        // A fragment: two contiguous 8-element bf16 runs from LDS
        const __bf16* ap = As + mn * DFEAT + k0 + half * 16;
        v8bf lo = *(const v8bf*)(ap);
        v8bf hi = *(const v8bf*)(ap + 8);
        v16bf a = __builtin_shufflevector(lo, hi,
            0, 1, 2, 3, 4, 5, 6, 7, 8, 9, 10, 11, 12, 13, 14, 15);
        // B fragment: 16 consecutive bf16 from transposed weights (32B load)
        const __bf16* bp = Wt + (size_t)(col0 + mn) * DFEAT + k0 + half * 16;
        v16bf b = *(const v16bf*)bp;

        acc = __builtin_amdgcn_wmma_f32_16x16x32_bf16(
            false, a, false, b, (short)0, acc, false, false);
    }

#pragma unroll
    for (int r = 0; r < 8; ++r) {
        const long row = row0 + r + 8 * half;
        const int  col = col0 + mn;
        float v = acc[r] + bias[col];
        if (act) v = gelu_exact(v);
        C[row * DFEAT + col] = v;
    }
}

// ---------------------------------------------------------------------------
// Skinny projection: out[N,4] = x[N,128] @ w[128,4] + b[4]
// ---------------------------------------------------------------------------
__global__ void attn_logits(const float* __restrict__ x, const float* __restrict__ w,
                            const float* __restrict__ b, float* __restrict__ out, int n)
{
    int i = blockIdx.x * blockDim.x + threadIdx.x;
    if (i >= n) return;
    const float* xr = x + (size_t)i * DFEAT;
    float acc[HN];
#pragma unroll
    for (int h = 0; h < HN; ++h) acc[h] = b[h];
    for (int k = 0; k < DFEAT; ++k) {
        const float xv = xr[k];
#pragma unroll
        for (int h = 0; h < HN; ++h) acc[h] += xv * w[k * HN + h];
    }
#pragma unroll
    for (int h = 0; h < HN; ++h) out[(size_t)i * HN + h] = acc[h];
}

// ---------------------------------------------------------------------------
// Edge softmax phase 1: leaky-relu logit + segment max (ordered-int atomicMax)
// ---------------------------------------------------------------------------
__global__ void edge_max(const float* __restrict__ aS, const float* __restrict__ aD,
                         const int* __restrict__ es, const int* __restrict__ ed,
                         float* __restrict__ aE, int* __restrict__ mOrd, int nE)
{
    int idx = blockIdx.x * blockDim.x + threadIdx.x;
    if (idx >= nE * HN) return;
    const int e = idx >> 2, h = idx & 3;
    const float v = lrelu(aS[(size_t)es[e] * HN + h] + aD[(size_t)ed[e] * HN + h]);
    aE[idx] = v;
    atomicMax(&mOrd[(size_t)ed[e] * HN + h], f2ord(v));
}

// Phase 2: exp(a - max) + segment sum
__global__ void edge_exp(float* __restrict__ aE, const int* __restrict__ ed,
                         const int* __restrict__ mOrd, float* __restrict__ sSum, int nE)
{
    int idx = blockIdx.x * blockDim.x + threadIdx.x;
    if (idx >= nE * HN) return;
    const int e = idx >> 2, h = idx & 3;
    const float m  = ord2f(mOrd[(size_t)ed[e] * HN + h]);
    const float ev = expf(aE[idx] - m);
    aE[idx] = ev;
    atomicAdd(&sSum[(size_t)ed[e] * HN + h], ev);
}

// Phase 3: z[dst] += hs[src] * attn  (thread per (edge, feature))
__global__ void edge_agg(const float* __restrict__ hs, const float* __restrict__ aE,
                         const float* __restrict__ sSum,
                         const int* __restrict__ es, const int* __restrict__ ed,
                         float* __restrict__ z, int nE)
{
    long idx = (long)blockIdx.x * blockDim.x + threadIdx.x;
    if (idx >= (long)nE * DFEAT) return;
    const int e = (int)(idx >> 7);
    const int f = (int)(idx & 127);
    const int head = f >> 5;            // D/HN = 32
    const int s = es[e], d = ed[e];
    const float a = aE[(size_t)e * HN + head] / sSum[(size_t)d * HN + head];
    atomicAdd(&z[(size_t)d * DFEAT + f], hs[(size_t)s * DFEAT + f] * a);
}

// ---------------------------------------------------------------------------
// Node-independent relation attention: softmax_r(leaky(rel_emb @ relA_w + b))
// ---------------------------------------------------------------------------
__global__ void rel_softmax(const float* __restrict__ re, const float* __restrict__ rw,
                            const float* __restrict__ rb, float* __restrict__ out, int R)
{
    __shared__ float lg[3][HT];
    const int tid = threadIdx.x, r = tid >> 2, t = tid & 3;
    if (r < R && tid < 12) {
        float acc = rb[r * HT + t];
        for (int k = 0; k < DFEAT; ++k)
            acc += re[r * DFEAT + k] * rw[r * DFEAT * HT + k * HT + t];
        lg[r][t] = lrelu(acc);
    }
    __syncthreads();
    if (tid < HT) {
        float m = -3.4e38f;
        for (int rr = 0; rr < R; ++rr) m = fmaxf(m, lg[rr][tid]);
        float e[3], s = 0.0f;
        for (int rr = 0; rr < R; ++rr) { e[rr] = expf(lg[rr][tid] - m); s += e[rr]; }
        for (int rr = 0; rr < R; ++rr) out[rr * HT + tid] = e[rr] / s;
    }
}

__device__ __forceinline__ float block_reduce_128(float v, float* red)
{
    const int tid = threadIdx.x;
    red[tid] = v;
    __syncthreads();
#pragma unroll
    for (int s = 64; s > 0; s >>= 1) {
        if (tid < s) red[tid] += red[tid + s];
        __syncthreads();
    }
    const float r = red[0];
    __syncthreads();
    return r;
}

// ---------------------------------------------------------------------------
// Semantic attention: one 128-thread block per node. Output overwrites hd
// in-place (each thread reads its own element before any write).
// ---------------------------------------------------------------------------
__global__ __launch_bounds__(128) void semantic_kernel(
    const float* z0, const float* z1, const float* z2, int R,
    const float* __restrict__ h, float* __restrict__ hd,
    const float* __restrict__ sSw, const float* __restrict__ sSb,
    const float* __restrict__ sDw, const float* __restrict__ sDb,
    const float* __restrict__ relW)
{
    __shared__ float red[128];
    __shared__ float dS[3][HT], dD[3][HT], attn[3][HT];
    const int tid = threadIdx.x;
    const size_t base = (size_t)blockIdx.x * DFEAT + tid;

    const float* zp[3] = {z0, z1, z2};
    float zr[3] = {0.0f, 0.0f, 0.0f};
    for (int r = 0; r < R; ++r) zr[r] = zp[r][base];
    const float hv  = h[base];
    const float hdv = hd[base];

    const float ssd = block_reduce_128(hdv * hdv, red);
    const float zdn = hdv * (1.0f / fmaxf(sqrtf(ssd), 1e-9f));

    float znr[3] = {0.0f, 0.0f, 0.0f};
    for (int r = 0; r < R; ++r) {
        const float ss = block_reduce_128(zr[r] * zr[r], red);
        znr[r] = zr[r] * (1.0f / fmaxf(sqrtf(ss), 1e-9f));
    }

    for (int r = 0; r < R; ++r)
        for (int t = 0; t < HT; ++t) {
            const float s1 = block_reduce_128(znr[r] * sSw[r * DFEAT * HT + tid * HT + t], red);
            const float s2 = block_reduce_128(zdn    * sDw[r * DFEAT * HT + tid * HT + t], red);
            if (tid == 0) {
                dS[r][t] = s1 + sSb[r * HT + t];
                dD[r][t] = s2 + sDb[r * HT + t];
            }
        }
    __syncthreads();

    if (tid < HT) {
        const int t = tid;
        float lg[3], m = -3.4e38f;
        for (int r = 0; r < R; ++r) { lg[r] = lrelu(dS[r][t] + dD[r][t]); m = fmaxf(m, lg[r]); }
        float se = 0.0f;
        for (int r = 0; r < R; ++r) { lg[r] = expf(lg[r] - m); se += lg[r]; }
        for (int r = 0; r < R; ++r) attn[r][t] = 0.5f * (lg[r] / se) + 0.5f * relW[r * HT + t];
    }
    __syncthreads();

    const int t = tid >> 5;   // D/HT = 32
    float acc = hv;
    for (int r = 0; r < R; ++r) acc += zr[r] * attn[r][t];
    const float g   = gelu_exact(acc);
    const float ssg = block_reduce_128(g * g, red);
    hd[base] = g * (1.0f / fmaxf(sqrtf(ssg), 1e-9f));
}

// ---------------------------------------------------------------------------
// Final link scores: out[0..P) = pos dots, out[P..2P) = neg dots
// ---------------------------------------------------------------------------
__global__ void pair_dot(const float* __restrict__ h,
                         const int* __restrict__ ps, const int* __restrict__ pd,
                         const int* __restrict__ ns, const int* __restrict__ nd,
                         float* __restrict__ out, int P)
{
    int i = blockIdx.x * blockDim.x + threadIdx.x;
    if (i >= P) return;
    {
        const float* a = h + (size_t)ps[i] * DFEAT;
        const float* b = h + (size_t)pd[i] * DFEAT;
        float s = 0.0f;
        for (int k = 0; k < DFEAT; ++k) s += a[k] * b[k];
        out[i] = s;
    }
    {
        const float* a = h + (size_t)ns[i] * DFEAT;
        const float* b = h + (size_t)nd[i] * DFEAT;
        float s = 0.0f;
        for (int k = 0; k < DFEAT; ++k) s += a[k] * b[k];
        out[P + i] = s;
    }
}

// ---------------------------------------------------------------------------
// Host orchestration
// ---------------------------------------------------------------------------
extern "C" void kernel_launch(void* const* d_in, const int* in_sizes, int n_in,
                              void* d_out, int out_size, void* d_ws, size_t ws_size,
                              hipStream_t stream)
{
    (void)in_sizes; (void)n_in; (void)out_size; (void)ws_size;

    const float* x_author  = (const float*)d_in[0];
    const float* x_paper   = (const float*)d_in[1];
    const int* writes_src  = (const int*)d_in[2];
    const int* writes_dst  = (const int*)d_in[3];
    const int* cites_src   = (const int*)d_in[4];
    const int* cites_dst   = (const int*)d_in[5];
    const int* pos_src     = (const int*)d_in[6];
    const int* pos_dst     = (const int*)d_in[7];
    const int* neg_src     = (const int*)d_in[8];
    const int* neg_dst     = (const int*)d_in[9];
    // params dict, insertion order:
    const float* pre_w  = (const float*)d_in[10];   // [2,128,128]
    const float* pre_b  = (const float*)d_in[11];   // [2,128]
    const float* node_w = (const float*)d_in[12];   // [2,2,128,128]
    const float* node_b = (const float*)d_in[13];   // [2,2,128]
    const float* edge_w = (const float*)d_in[14];   // [2,4,128,128]
    const float* edge_b = (const float*)d_in[15];   // [2,4,128]
    const float* srcA_w = (const float*)d_in[16];   // [2,4,128,4]
    const float* srcA_b = (const float*)d_in[17];   // [2,4,4]
    const float* dstA_w = (const float*)d_in[18];
    const float* dstA_b = (const float*)d_in[19];
    const float* semS_w = (const float*)d_in[20];   // [2,4,128,4]
    const float* semS_b = (const float*)d_in[21];
    const float* semD_w = (const float*)d_in[22];
    const float* semD_b = (const float*)d_in[23];
    const float* relA_w = (const float*)d_in[24];
    const float* relA_b = (const float*)d_in[25];
    const float* rel_emb = (const float*)d_in[26];  // [2,4,128]

    // ---- workspace carve (requires ~480 MB) ----
    char* wsp = (char*)d_ws;
    auto take = [&](size_t bytes) -> float* {
        char* p = wsp;
        wsp += (bytes + 255) & ~(size_t)255;
        return (float*)p;
    };
    const size_t NODEB = (size_t)NAUT * DFEAT * sizeof(float);
    float* a0 = take(NODEB);   // author h (ping-pong)
    float* a1 = take(NODEB);
    float* p0 = take(NODEB);   // paper h (ping-pong)
    float* p1 = take(NODEB);
    float* hs = take(NODEB);   // edge-transformed source features
    float* z[4];
    for (int i = 0; i < 4; ++i) z[i] = take(NODEB);
    float* aSrc = take((size_t)NAUT * HN * sizeof(float));
    float* aDst = take((size_t)NAUT * HN * sizeof(float));
    int*   mOrd = (int*)take((size_t)NAUT * HN * sizeof(int));
    float* sSum = take((size_t)NAUT * HN * sizeof(float));
    float* aE   = take((size_t)NEDGE * HN * sizeof(float));
    float* relW = take(64 * sizeof(float));
    __bf16* wT  = (__bf16*)take((size_t)DFEAT * DFEAT * sizeof(__bf16)); // 32KB

    const int N = NAUT;                 // NA == NP == 100000
    const int gemmGrid = N / 16;        // 6250 (exact)
    const int convGrid = (DFEAT * DFEAT) / 256;   // 64
    const int nodeHGrid = (N * HN + 255) / 256;
    const int nodeFGrid = (int)(((long)N * DFEAT + 255) / 256);
    const int edgeHGrid = (NEDGE * HN + 255) / 256;
    const int edgeFGrid = (int)(((long)NEDGE * DFEAT + 255) / 256);

    // helper: prep weights then run the WMMA GEMM
    auto gemm = [&](const float* A, const float* W, const float* b,
                    float* C, int act) {
        conv_w_t<<<convGrid, 256, 0, stream>>>(W, wT);
        gemm128_wmma<<<gemmGrid, 256, 0, stream>>>(A, wT, b, C, act);
    };

    // ---- pre-projection: h = gelu(x @ pre_w + pre_b) ----
    gemm(x_author, pre_w, pre_b, a0, 1);
    gemm(x_paper, pre_w + DFEAT * DFEAT, pre_b + DFEAT, p0, 1);

    for (int l = 0; l < 2; ++l) {
        // node transforms hd = h @ node_w[l,t] + node_b[l,t]
        gemm(a0, node_w + (size_t)(l * 2 + 0) * DFEAT * DFEAT,
             node_b + (size_t)(l * 2 + 0) * DFEAT, a1, 0);
        gemm(p0, node_w + (size_t)(l * 2 + 1) * DFEAT * DFEAT,
             node_b + (size_t)(l * 2 + 1) * DFEAT, p1, 0);

        // relations: writes, written_by, cites, cited_by
        const float* relSrcH[4] = {a0, p0, p0, p0};
        const float* relDstHd[4] = {p1, a1, p1, p1};
        const int* relEs[4] = {writes_src, writes_dst, cites_src, cites_dst};
        const int* relEd[4] = {writes_dst, writes_src, cites_dst, cites_src};

        for (int e = 0; e < 4; ++e) {
            const size_t we = (size_t)(l * 4 + e);
            // hs = h_src @ edge_w + edge_b
            gemm(relSrcH[e], edge_w + we * DFEAT * DFEAT, edge_b + we * DFEAT, hs, 0);
            // attention logits
            attn_logits<<<(N + 127) / 128, 128, 0, stream>>>(
                hs, srcA_w + we * DFEAT * HN, srcA_b + we * HN, aSrc, N);
            attn_logits<<<(N + 127) / 128, 128, 0, stream>>>(
                relDstHd[e], dstA_w + we * DFEAT * HN, dstA_b + we * HN, aDst, N);
            // edge softmax + aggregate
            fill_i32<<<nodeHGrid, 256, 0, stream>>>(mOrd, (int)0x80000000, (long)N * HN);
            fill_f32<<<nodeHGrid, 256, 0, stream>>>(sSum, 0.0f, (long)N * HN);
            fill_f32<<<nodeFGrid, 256, 0, stream>>>(z[e], 0.0f, (long)N * DFEAT);
            edge_max<<<edgeHGrid, 256, 0, stream>>>(aSrc, aDst, relEs[e], relEd[e],
                                                    aE, mOrd, NEDGE);
            edge_exp<<<edgeHGrid, 256, 0, stream>>>(aE, relEd[e], mOrd, sSum, NEDGE);
            edge_agg<<<edgeFGrid, 256, 0, stream>>>(hs, aE, sSum, relEs[e], relEd[e],
                                                    z[e], NEDGE);
        }

        // semantic attention: author (aidx base 0, R=1 with z_wb)
        {
            const size_t ai = (size_t)(l * 4 + 0);
            rel_softmax<<<1, 32, 0, stream>>>(rel_emb + ai * DFEAT,
                                              relA_w + ai * DFEAT * HT,
                                              relA_b + ai * HT, relW, 1);
            semantic_kernel<<<N, 128, 0, stream>>>(
                z[1], z[1], z[1], 1, a0, a1,
                semS_w + ai * DFEAT * HT, semS_b + ai * HT,
                semD_w + ai * DFEAT * HT, semD_b + ai * HT, relW);
        }
        // paper (aidx base 1, R=3 with z_wr, z_ci, z_cb)
        {
            const size_t ai = (size_t)(l * 4 + 1);
            rel_softmax<<<1, 32, 0, stream>>>(rel_emb + ai * DFEAT,
                                              relA_w + ai * DFEAT * HT,
                                              relA_b + ai * HT, relW, 3);
            semantic_kernel<<<N, 128, 0, stream>>>(
                z[0], z[2], z[3], 3, p0, p1,
                semS_w + ai * DFEAT * HT, semS_b + ai * HT,
                semD_w + ai * DFEAT * HT, semD_b + ai * HT, relW);
        }

        // ping-pong: new h lives in a1/p1
        float* t;
        t = a0; a0 = a1; a1 = t;
        t = p0; p0 = p1; p1 = t;
    }

    pair_dot<<<(NPAIR + 255) / 256, 256, 0, stream>>>(
        a0, pos_src, pos_dst, neg_src, neg_dst, (float*)d_out, NPAIR);
}
